// MoE_layer_megatron_wo_gate_v2_53601191854776
// MI455X (gfx1250) — compile-verified, matrix operand order checked by reading, not run on hardware
//
#include <hip/hip_runtime.h>
#include <hip/hip_bf16.h>
#include <math.h>

// Problem constants (fixed by the reference).
#define T_TOK 4096
#define DDIM  1024
#define FDIM  2048
#define NEXP  8
#define TOPKK 5
#define NPAIR (T_TOK * TOPKK)   // 20480 dispatched (token, slot) pairs

// Workspace layout (bytes)
#define WS_COUNTS 0
#define WS_OFFS   32
#define WS_CURS   64
#define WS_PAIRS  128                                   // NPAIR int (~80 KB)
#define WS_XB     131072                                // x  bf16: T*D*2        = 8 MB
#define WS_W1B    (WS_XB  + (size_t)T_TOK * DDIM * 2)   // w1^T bf16 [E][F][D]   = 32 MB
#define WS_W2B    (WS_W1B + (size_t)NEXP * DDIM * FDIM * 2) // w2^T bf16 [E][D][F] = 32 MB
#define WS_H      (WS_W2B + (size_t)NEXP * FDIM * DDIM * 2) // h bf16 [NPAIR][F]  = 80 MB

typedef __attribute__((ext_vector_type(16))) __bf16 v16bf;
typedef __attribute__((ext_vector_type(8)))  float  v8f;
typedef __attribute__((ext_vector_type(4)))  unsigned int u32x4;
typedef __attribute__((ext_vector_type(8)))  int i32x8;
typedef __attribute__((ext_vector_type(4)))  int i32x4;

// ---------- helpers ----------

static __device__ inline unsigned pack_bf2(float a, float b) {
  unsigned short lo = __builtin_bit_cast(unsigned short, (__bf16)a);
  unsigned short hi = __builtin_bit_cast(unsigned short, (__bf16)b);
  return (unsigned)lo | ((unsigned)hi << 16);
}

static __device__ inline unsigned short bf16_bits(float a) {
  return __builtin_bit_cast(unsigned short, (__bf16)a);
}

static __device__ inline float gelu_tanh(float v) {
  // jax.nn.gelu default (approximate=True); branch-free tanh via fast exp.
  const float c = 0.7978845608028654f;   // sqrt(2/pi)
  float u = c * (v + 0.044715f * v * v * v);
  float t = 1.0f - 2.0f / (__expf(2.0f * u) + 1.0f);
  return 0.5f * v * (1.0f + t);
}

// Load one 16x32 bf16 WMMA operand fragment from an LDS tile with row stride
// 64 elements (K-step 64, two K-halves per row).
// Per CDNA5 ISA 16-bit operand layout: lane l holds row (l&15);
// lanes 0-15 take K {0..7, 16..23} of the half, lanes 16-31 take {8..15, 24..31}:
// two aligned 16-byte runs per lane -> two ds_load_b128.
static __device__ inline v16bf load_frag64(const unsigned short* s, int row,
                                           int khalf, int lane) {
  int base = row * 64 + khalf * 32 + ((lane & 16) ? 8 : 0);
  union { uint4 q[2]; v16bf v; } u;
  u.q[0] = *(const uint4*)(s + base);
  u.q[1] = *(const uint4*)(s + base + 16);
  return u.v;
}

// Tensor Data Mover: DMA one 2D bf16 tile (tile_k x tile_n rows) from global
// into LDS at lds_off, rows packed at stride tile_k elements.
// D# bitfields per CDNA5 ISA.
static __device__ inline void tdm_load_2d(unsigned lds_off, const void* gaddr,
                                          unsigned tile_k, unsigned tile_n,
                                          unsigned rem_k, unsigned rem_n,
                                          unsigned stride_k /*elements*/) {
  unsigned long long ga = (unsigned long long)(size_t)gaddr;
  u32x4 g0;
  g0.x = 1u;                                         // count=1, user descriptor
  g0.y = lds_off;                                    // lds_addr (bytes)
  g0.z = (unsigned)(ga & 0xffffffffu);               // global_addr[31:0]
  g0.w = (unsigned)((ga >> 32) & 0x01ffffffu)        // global_addr[56:32]
         | (2u << 30);                               // type=2 ("image")
  i32x8 g1;
  g1[0] = (int)(1u << 16);                           // data_size=2B; mask/flags 0
  g1[1] = (int)((rem_k & 0xffffu) << 16);            // tensor_dim0[15:0]
  g1[2] = (int)(((rem_k >> 16) & 0xffffu)            // tensor_dim0[31:16]
                | ((rem_n & 0xffffu) << 16));        // tensor_dim1[15:0]
  g1[3] = (int)(((rem_n >> 16) & 0xffffu)            // tensor_dim1[31:16]
                | ((tile_k & 0xffffu) << 16));       // tile_dim0
  g1[4] = (int)(tile_n & 0xffffu);                   // tile_dim1 (tile_dim2=0)
  g1[5] = (int)stride_k;                             // tensor_dim0_stride[31:0]
  g1[6] = 0;                                         // stride hi / dim1_stride lo
  g1[7] = 0;
  i32x4 z4 = {0, 0, 0, 0};
#if __clang_major__ >= 23
  i32x8 z8 = {0, 0, 0, 0, 0, 0, 0, 0};
  __builtin_amdgcn_tensor_load_to_lds(g0, g1, z4, z4, z8, 0);
#else
  __builtin_amdgcn_tensor_load_to_lds(g0, g1, z4, z4, 0);
#endif
}

// ---------- dispatch / conversion kernels ----------

__global__ void k_zero_out(float4* out4) {
  int i = blockIdx.x * blockDim.x + threadIdx.x;  // grid sized to T*D/4
  out4[i] = make_float4(0.f, 0.f, 0.f, 0.f);
}

__global__ void k_init_meta(int* ws_ints) {
  int i = threadIdx.x;
  if (i < 24) ws_ints[i] = 0;
}

__global__ void k_count(const int* __restrict__ ce, int* __restrict__ counts) {
  int i = blockIdx.x * blockDim.x + threadIdx.x;
  if (i < NPAIR) atomicAdd(&counts[ce[i]], 1);
}

__global__ void k_scan(const int* __restrict__ counts,
                       int* __restrict__ offs, int* __restrict__ curs) {
  if (threadIdx.x == 0) {
    int a = 0;
    for (int e = 0; e < NEXP; ++e) { offs[e] = a; curs[e] = a; a += counts[e]; }
  }
}

__global__ void k_scatter(const int* __restrict__ ce,
                          int* __restrict__ curs, int* __restrict__ pairs) {
  int i = blockIdx.x * blockDim.x + threadIdx.x;
  if (i < NPAIR) {
    int pos = atomicAdd(&curs[ce[i]], 1);
    pairs[pos] = i;               // i = t*TOPK + slot
  }
}

// x (f32) -> bf16, elementwise; each thread handles 8 values.
__global__ void k_cvt_x(const float* __restrict__ in, unsigned short* __restrict__ out) {
  int i = blockIdx.x * blockDim.x + threadIdx.x;
  const float4* p = (const float4*)(in + (size_t)i * 8);
  float4 f0 = p[0], f1 = p[1];
  uint4 q;
  q.x = pack_bf2(f0.x, f0.y); q.y = pack_bf2(f0.z, f0.w);
  q.z = pack_bf2(f1.x, f1.y); q.w = pack_bf2(f1.z, f1.w);
  *(uint4*)(out + (size_t)i * 8) = q;
}

// Transpose + convert: per expert, in[R][C] f32 -> out[C][R] bf16.
// 32x32 LDS tile; coalesced read along C, coalesced write along R.
__global__ __launch_bounds__(256)
void k_tcvt(const float* __restrict__ in, unsigned short* __restrict__ out,
            int R, int C) {
  __shared__ float tile[32 * 33];
  const int e  = blockIdx.z;
  const int r0 = blockIdx.y * 32;
  const int c0 = blockIdx.x * 32;
  const float* ine = in + (size_t)e * R * C;
  unsigned short* oute = out + (size_t)e * R * C;

  const int tid = threadIdx.x;
  {
    int r = tid >> 3;            // 0..31
    int c = (tid & 7) * 4;       // 0..28
    float4 f = *(const float4*)(ine + (size_t)(r0 + r) * C + c0 + c);
    tile[r * 33 + c + 0] = f.x;
    tile[r * 33 + c + 1] = f.y;
    tile[r * 33 + c + 2] = f.z;
    tile[r * 33 + c + 3] = f.w;
  }
  __syncthreads();
  {
    int c = tid >> 3;            // 0..31
    int r = (tid & 7) * 4;       // 0..28
    uint2 q;
    q.x = pack_bf2(tile[(r + 0) * 33 + c], tile[(r + 1) * 33 + c]);
    q.y = pack_bf2(tile[(r + 2) * 33 + c], tile[(r + 3) * 33 + c]);
    *(uint2*)(oute + (size_t)(c0 + c) * R + r0 + r) = q;
  }
}

// ---------- grouped GEMMs (bf16 WMMA, f32 accumulate) ----------
// Block tile 64(M) x 128(N), K-step 64. 8 wave32 waves as 4(M) x 2(N);
// each wave owns a 16x64 strip (4 accumulators, A-fragment reuse x4).
// Double-buffered software pipeline: while iteration i computes on buffer b,
// the TDM streams the next B tile into sB[1-b] and all threads copy the next
// A tile into sA[1-b]. One barrier per iteration:
//  - entering iter i implies every wave finished reading the 1-b buffers
//    (each wave's dscnt wait precedes its barrier signal), so overwriting is safe;
//  - wave 0 waits TENSORcnt before signaling, so the barrier publishes sB.

#define BK 64

__global__ __launch_bounds__(256)
void k_ffn1(const unsigned short* __restrict__ xb, const unsigned short* __restrict__ w1b,
            const int* __restrict__ counts, const int* __restrict__ offs,
            const int* __restrict__ pairs, unsigned short* __restrict__ h) {
  const int e   = blockIdx.z;
  const int cnt = counts[e];
  const int mt  = blockIdx.x;
  if (mt * 64 >= cnt) return;
  const int off = offs[e];
  const int nb  = blockIdx.y * 128;

  __shared__ unsigned short sA[2][64 * BK];    // [m][k] bf16
  __shared__ unsigned short sB[2][128 * BK];   // [n][k] bf16, filled by TDM

  const int tid  = threadIdx.x;
  const int lane = tid & 31;
  const int wave = tid >> 5;
  const int wm   = wave & 3;
  const int wn   = wave >> 2;

  // A-loader: thread -> (row 0..63, 16-element k-chunk)
  const int ar  = tid >> 2;
  const int akc = (tid & 3) * 16;
  const int rmax = off + cnt - 1;
  int rowg = off + mt * 64 + ar;
  if (rowg > rmax) rowg = rmax;
  const unsigned short* arow = xb + (size_t)(pairs[rowg] / TOPKK) * DDIM;

  // Uniform base of this block's B panel: w1^T[e][nb..nb+127][*]
  const unsigned short* bpanel = w1b + (size_t)e * FDIM * DDIM + (size_t)nb * DDIM;
  const unsigned sB_off[2] = { (unsigned)(size_t)&sB[0][0],
                               (unsigned)(size_t)&sB[1][0] };

  v8f acc[4] = {};

  // prologue: fill buffer 0
  {
    uint4 a0 = *(const uint4*)(arow + akc);
    uint4 a1 = *(const uint4*)(arow + akc + 8);
    *(uint4*)(&sA[0][ar * BK + akc])     = a0;
    *(uint4*)(&sA[0][ar * BK + akc + 8]) = a1;
    if (wave == 0) {
      tdm_load_2d(sB_off[0], bpanel, BK, 128, DDIM, 128, DDIM);
      __builtin_amdgcn_s_wait_tensorcnt(0);
    }
  }
  __syncthreads();

  const int nsteps = DDIM / BK;
  for (int i = 0; i < nsteps; ++i) {
    const int b = i & 1;
    if (i + 1 < nsteps) {
      const int kn = (i + 1) * BK;
      uint4 a0 = *(const uint4*)(arow + kn + akc);
      uint4 a1 = *(const uint4*)(arow + kn + akc + 8);
      *(uint4*)(&sA[1 - b][ar * BK + akc])     = a0;
      *(uint4*)(&sA[1 - b][ar * BK + akc + 8]) = a1;
      if (wave == 0)
        tdm_load_2d(sB_off[1 - b], bpanel + kn, BK, 128, DDIM - kn, 128, DDIM);
      if (i + 2 < nsteps)
        __builtin_prefetch(arow + (i + 2) * BK + akc, 0, 0);
    }

#pragma unroll
    for (int kh = 0; kh < 2; ++kh) {
      v16bf a  = load_frag64(sA[b], wm * 16 + (lane & 15), kh, lane);
      v16bf b0 = load_frag64(sB[b], wn * 64 +  0 + (lane & 15), kh, lane);
      v16bf b1 = load_frag64(sB[b], wn * 64 + 16 + (lane & 15), kh, lane);
      v16bf b2 = load_frag64(sB[b], wn * 64 + 32 + (lane & 15), kh, lane);
      v16bf b3 = load_frag64(sB[b], wn * 64 + 48 + (lane & 15), kh, lane);
      acc[0] = __builtin_amdgcn_wmma_f32_16x16x32_bf16(false, a, false, b0, (short)0, acc[0], false, false);
      acc[1] = __builtin_amdgcn_wmma_f32_16x16x32_bf16(false, a, false, b1, (short)0, acc[1], false, false);
      acc[2] = __builtin_amdgcn_wmma_f32_16x16x32_bf16(false, a, false, b2, (short)0, acc[2], false, false);
      acc[3] = __builtin_amdgcn_wmma_f32_16x16x32_bf16(false, a, false, b3, (short)0, acc[3], false, false);
    }

    if (wave == 0 && i + 1 < nsteps) __builtin_amdgcn_s_wait_tensorcnt(0);
    __syncthreads();
  }

  // epilogue: gelu + bf16 store of h
  const int mh = (lane & 16) ? 8 : 0;
  const int nl = lane & 15;
  const int mbase = mt * 64 + wm * 16 + mh;
#pragma unroll
  for (int j = 0; j < 4; ++j) {
    const int col = nb + wn * 64 + j * 16 + nl;
#pragma unroll
    for (int r = 0; r < 8; ++r) {
      int ml = mbase + r;
      if (ml < cnt)
        h[(size_t)(off + ml) * FDIM + col] = bf16_bits(gelu_tanh(acc[j][r]));
    }
  }
}

__global__ __launch_bounds__(256)
void k_ffn2(const unsigned short* __restrict__ h, const unsigned short* __restrict__ w2b,
            const float* __restrict__ gate, const int* __restrict__ counts,
            const int* __restrict__ offs, const int* __restrict__ pairs,
            float* __restrict__ out) {
  const int e   = blockIdx.z;
  const int cnt = counts[e];
  const int mt  = blockIdx.x;
  if (mt * 64 >= cnt) return;
  const int off = offs[e];
  const int nb  = blockIdx.y * 128;

  __shared__ unsigned short sA[2][64 * BK];
  __shared__ unsigned short sB[2][128 * BK];

  const int tid  = threadIdx.x;
  const int lane = tid & 31;
  const int wave = tid >> 5;
  const int wm   = wave & 3;
  const int wn   = wave >> 2;

  const int ar  = tid >> 2;
  const int akc = (tid & 3) * 16;
  const int rmax = off + cnt - 1;
  int rowg = off + mt * 64 + ar;
  if (rowg > rmax) rowg = rmax;
  const unsigned short* arow = h + (size_t)rowg * FDIM;

  const unsigned short* bpanel = w2b + (size_t)e * DDIM * FDIM + (size_t)nb * FDIM;
  const unsigned sB_off[2] = { (unsigned)(size_t)&sB[0][0],
                               (unsigned)(size_t)&sB[1][0] };

  v8f acc[4] = {};

  {
    uint4 a0 = *(const uint4*)(arow + akc);
    uint4 a1 = *(const uint4*)(arow + akc + 8);
    *(uint4*)(&sA[0][ar * BK + akc])     = a0;
    *(uint4*)(&sA[0][ar * BK + akc + 8]) = a1;
    if (wave == 0) {
      tdm_load_2d(sB_off[0], bpanel, BK, 128, FDIM, 128, FDIM);
      __builtin_amdgcn_s_wait_tensorcnt(0);
    }
  }
  __syncthreads();

  const int nsteps = FDIM / BK;
  for (int i = 0; i < nsteps; ++i) {
    const int b = i & 1;
    if (i + 1 < nsteps) {
      const int kn = (i + 1) * BK;
      uint4 a0 = *(const uint4*)(arow + kn + akc);
      uint4 a1 = *(const uint4*)(arow + kn + akc + 8);
      *(uint4*)(&sA[1 - b][ar * BK + akc])     = a0;
      *(uint4*)(&sA[1 - b][ar * BK + akc + 8]) = a1;
      if (wave == 0)
        tdm_load_2d(sB_off[1 - b], bpanel + kn, BK, 128, FDIM - kn, 128, FDIM);
      if (i + 2 < nsteps)
        __builtin_prefetch(arow + (i + 2) * BK + akc, 0, 0);
    }

#pragma unroll
    for (int kh = 0; kh < 2; ++kh) {
      v16bf a  = load_frag64(sA[b], wm * 16 + (lane & 15), kh, lane);
      v16bf b0 = load_frag64(sB[b], wn * 64 +  0 + (lane & 15), kh, lane);
      v16bf b1 = load_frag64(sB[b], wn * 64 + 16 + (lane & 15), kh, lane);
      v16bf b2 = load_frag64(sB[b], wn * 64 + 32 + (lane & 15), kh, lane);
      v16bf b3 = load_frag64(sB[b], wn * 64 + 48 + (lane & 15), kh, lane);
      acc[0] = __builtin_amdgcn_wmma_f32_16x16x32_bf16(false, a, false, b0, (short)0, acc[0], false, false);
      acc[1] = __builtin_amdgcn_wmma_f32_16x16x32_bf16(false, a, false, b1, (short)0, acc[1], false, false);
      acc[2] = __builtin_amdgcn_wmma_f32_16x16x32_bf16(false, a, false, b2, (short)0, acc[2], false, false);
      acc[3] = __builtin_amdgcn_wmma_f32_16x16x32_bf16(false, a, false, b3, (short)0, acc[3], false, false);
    }

    if (wave == 0 && i + 1 < nsteps) __builtin_amdgcn_s_wait_tensorcnt(0);
    __syncthreads();
  }

  // epilogue: per-lane row metadata, then gate-weighted atomic scatter
  const int mh = (lane & 16) ? 8 : 0;
  const int nl = lane & 15;
  const int mbase = mt * 64 + wm * 16 + mh;

  int   tokr[8];
  float gater[8];
  bool  valr[8];
#pragma unroll
  for (int r = 0; r < 8; ++r) {
    int ml = mbase + r;
    valr[r] = (ml < cnt);
    int rg = off + (valr[r] ? ml : (cnt - 1));
    int p  = pairs[rg];
    tokr[r]  = p / TOPKK;
    gater[r] = gate[p];
  }

#pragma unroll
  for (int j = 0; j < 4; ++j) {
    const int col = nb + wn * 64 + j * 16 + nl;
#pragma unroll
    for (int r = 0; r < 8; ++r) {
      if (valr[r])
        unsafeAtomicAdd(&out[(size_t)tokr[r] * DDIM + col], gater[r] * acc[j][r]);
    }
  }
}

// ---------- launch ----------

extern "C" void kernel_launch(void* const* d_in, const int* in_sizes, int n_in,
                              void* d_out, int out_size, void* d_ws, size_t ws_size,
                              hipStream_t stream) {
  const float* x    = (const float*)d_in[0];
  const float* gate = (const float*)d_in[1];
  const float* w1   = (const float*)d_in[2];
  const float* w2   = (const float*)d_in[3];
  const int*   ce   = (const int*)d_in[4];
  float*       out  = (float*)d_out;

  char* ws = (char*)d_ws;
  int* counts = (int*)(ws + WS_COUNTS);
  int* offs   = (int*)(ws + WS_OFFS);
  int* curs   = (int*)(ws + WS_CURS);
  int* pairs  = (int*)(ws + WS_PAIRS);
  unsigned short* xb  = (unsigned short*)(ws + WS_XB);
  unsigned short* w1b = (unsigned short*)(ws + WS_W1B);
  unsigned short* w2b = (unsigned short*)(ws + WS_W2B);
  unsigned short* h   = (unsigned short*)(ws + WS_H);

  // dispatch metadata
  k_zero_out<<<(T_TOK * DDIM / 4) / 256, 256, 0, stream>>>((float4*)out);
  k_init_meta<<<1, 32, 0, stream>>>((int*)ws);
  k_count<<<NPAIR / 256, 256, 0, stream>>>(ce, counts);
  k_scan<<<1, 32, 0, stream>>>(counts, offs, curs);
  k_scatter<<<NPAIR / 256, 256, 0, stream>>>(ce, curs, pairs);

  // precision/layout conversion: x -> bf16; w1 -> [E][F][D] bf16; w2 -> [E][D][F] bf16
  k_cvt_x<<<(T_TOK * DDIM / 8) / 256, 256, 0, stream>>>(x, xb);
  dim3 gt1(FDIM / 32, DDIM / 32, NEXP);
  k_tcvt<<<gt1, 256, 0, stream>>>(w1, w1b, DDIM, FDIM);
  dim3 gt2(DDIM / 32, FDIM / 32, NEXP);
  k_tcvt<<<gt2, 256, 0, stream>>>(w2, w2b, FDIM, DDIM);

  // grouped GEMMs
  dim3 g1((NPAIR + 63) / 64, FDIM / 128, NEXP);
  k_ffn1<<<g1, 256, 0, stream>>>(xb, w1b, counts, offs, pairs, h);

  dim3 g2((NPAIR + 63) / 64, DDIM / 128, NEXP);
  k_ffn2<<<g2, 256, 0, stream>>>(h, w2b, gate, counts, offs, pairs, out);
}
